// MultiHeadedAttention_26293789786444
// MI455X (gfx1250) — matrix operand order repaired and checked by reading, not compiled, not run on hardware
//
#include <hip/hip_runtime.h>
#include <stdint.h>

typedef __attribute__((ext_vector_type(16))) __bf16 v16bf;
typedef __attribute__((ext_vector_type(8)))  float  v8f;
typedef __attribute__((ext_vector_type(4)))  unsigned int v4u;
typedef __attribute__((ext_vector_type(8)))  int v8i;
typedef __attribute__((ext_vector_type(4)))  int v4i;

#define B_ 2
#define T_ 2048
#define C_ 1024
#define H_ 16
#define D_ 64
#define M_ (B_*T_)   // 4096 rows of x

#ifdef __has_builtin
#if __has_builtin(__builtin_amdgcn_tensor_load_to_lds) && __has_builtin(__builtin_amdgcn_s_wait_tensorcnt)
#define HAVE_TDM 1
#endif
#endif
#ifndef HAVE_TDM
#define HAVE_TDM 0
#endif

__device__ __forceinline__ unsigned short f2bf(float f) {
  unsigned int u = __float_as_uint(f);
  u += 0x7FFFu + ((u >> 16) & 1u);   // round-to-nearest-even
  return (unsigned short)(u >> 16);
}

union Frag16 { v16bf v; uint4 q[2]; unsigned short s[16]; };
union Pack8  { uint4 q; unsigned short s[8]; };

// ---------------------------------------------------------------- zero OSUM
__global__ void mha_zero(float* p, int n) {
  int i = blockIdx.x * blockDim.x + threadIdx.x;
  if (i < n) p[i] = 0.f;
}

// ------------------------------------------- fp32 -> bf16 (+ W transposes)
__global__ void mha_convert(const float* __restrict__ x,
                            const float* __restrict__ wq,
                            const float* __restrict__ wk,
                            const float* __restrict__ wv,
                            unsigned short* __restrict__ XB,
                            unsigned short* __restrict__ WQT,
                            unsigned short* __restrict__ WKT,
                            unsigned short* __restrict__ WVT) {
  long i = (long)blockIdx.x * blockDim.x + threadIdx.x;
  const long NX = (long)M_ * C_;
  const long NW = (long)C_ * C_;
  if (i < NX) { XB[i] = f2bf(x[i]); return; }
  i -= NX;
  if (i < NW) { long n = i / C_, k = i % C_; WQT[i] = f2bf(wq[k * C_ + n]); return; }
  i -= NW;
  if (i < NW) { long n = i / C_, k = i % C_; WKT[i] = f2bf(wk[k * C_ + n]); return; }
  i -= NW;
  if (i < NW) { long n = i / C_, k = i % C_; WVT[i] = f2bf(wv[k * C_ + n]); return; }
}

// ---------------------------------------------------- QKV GEMM (bf16 WMMA)
// grid: (M/64, H, 3)  block: 128 (4 waves, each owns a 16-wide N strip)
// A (x) tile is staged 64x32 into LDS by the Tensor Data Mover, double
// buffered; B (weight^T) streams from global (L2-resident, shared by all
// M-tiles).
__global__ void mha_qkv(const unsigned short* __restrict__ XB,
                        const unsigned short* __restrict__ WQT,
                        const unsigned short* __restrict__ WKT,
                        const unsigned short* __restrict__ WVT,
                        unsigned short* __restrict__ Qb,   // [B][H][T][D]
                        unsigned short* __restrict__ Kb,   // [B][H][T][D]
                        unsigned short* __restrict__ VT) { // [B][H][D][T]
  __shared__ __align__(16) unsigned short Xs[2][64 * 32];
  const int wave = threadIdx.x >> 5;
  const int lane = threadIdx.x & 31;
  const int lo = lane & 15, hi = lane >> 4;
  const int m0 = blockIdx.x * 64;
  const int h  = blockIdx.y;               // each 64-wide N tile == one head
  const int n0 = h * 64 + wave * 16;
  const int z  = blockIdx.z;
  const unsigned short* WT = (z == 0) ? WQT : (z == 1) ? WKT : WVT;

  v8f zero = {};
  v8f acc[4] = {zero, zero, zero, zero};

#if HAVE_TDM
  // --- TDM descriptor builder: load 64(rows) x 32(cols) bf16 tile of XB ---
  auto tdm_issue = [&](int buf, int kk) {
    size_t ga = (size_t)(const void*)XB + ((size_t)m0 * C_ + kk) * 2;
    unsigned lds = (unsigned)(uintptr_t)(void*)&Xs[buf][0];
    v4u g0;
    g0.x = 1u;                                   // count=1 valid descriptor
    g0.y = lds;                                  // lds_addr [63:32]
    g0.z = (unsigned)ga;                         // global_addr low
    g0.w = ((unsigned)(ga >> 32) & 0x01FFFFFFu)  // global_addr[56:32]
           | 0x80000000u;                        // type=2 ("image")
    v8i g1;
    g1[0] = 0x00010000;           // workgroup_mask=0, data_size=1 (2 bytes)
    g1[1] = (int)(1024u << 16);   // tensor_dim0[15:0]=C_=1024
    g1[2] = (int)(4096u << 16);   // tensor_dim0 hi=0, tensor_dim1[15:0]=M_
    g1[3] = (int)(32u << 16);     // tensor_dim1 hi=0, tile_dim0=32
    g1[4] = 64;                   // tile_dim1=64, tile_dim2=0
    g1[5] = 1024;                 // tensor_dim0_stride low32 = C_
    g1[6] = 0;                    // stride0 hi, tensor_dim1_stride low16
    g1[7] = 0;                    // tensor_dim1_stride hi
    v4i gz4 = {0, 0, 0, 0};
    v8i gz8 = {0, 0, 0, 0, 0, 0, 0, 0};
    __builtin_amdgcn_tensor_load_to_lds(g0, g1, gz4, gz4, gz8, 0);
  };

  if (threadIdx.x < 32) tdm_issue(0, 0);
  int it = 0;
  for (int kk = 0; kk < C_; kk += 32, ++it) {
    if (threadIdx.x < 32) {
      if (kk + 32 < C_) {
        tdm_issue((it + 1) & 1, kk + 32);
        __builtin_amdgcn_s_wait_tensorcnt(1);   // current buffer resident
      } else {
        __builtin_amdgcn_s_wait_tensorcnt(0);
      }
    }
    __syncthreads();
    Frag16 bf;
    const unsigned short* bp = WT + (long)(n0 + lo) * C_ + kk + hi * 8;
    __builtin_prefetch(bp + 32, 0, 0);          // next K-step of this row
    bf.q[0] = *(const uint4*)(bp);
    bf.q[1] = *(const uint4*)(bp + 16);
    const unsigned short* xs = &Xs[it & 1][0];
    for (int i = 0; i < 4; i++) {
      Frag16 af;
      const unsigned short* ap = xs + (i * 16 + lo) * 32 + hi * 8;
      af.q[0] = *(const uint4*)(ap);
      af.q[1] = *(const uint4*)(ap + 16);
      acc[i] = __builtin_amdgcn_wmma_f32_16x16x32_bf16(
          false, af.v, false, bf.v, (short)0, acc[i], false, false);
    }
    __syncthreads();
  }
#else
  for (int kk = 0; kk < C_; kk += 32) {
    Frag16 bf;
    const unsigned short* bp = WT + (long)(n0 + lo) * C_ + kk + hi * 8;
    bf.q[0] = *(const uint4*)(bp);
    bf.q[1] = *(const uint4*)(bp + 16);
    for (int i = 0; i < 4; i++) {
      Frag16 af;
      const unsigned short* ap = XB + (long)(m0 + i * 16 + lo) * C_ + kk + hi * 8;
      af.q[0] = *(const uint4*)(ap);
      af.q[1] = *(const uint4*)(ap + 16);
      acc[i] = __builtin_amdgcn_wmma_f32_16x16x32_bf16(
          false, af.v, false, bf.v, (short)0, acc[i], false, false);
    }
  }
#endif

  const int dim = wave * 16 + lo;
  for (int i = 0; i < 4; i++) {
    int mrow0 = m0 + i * 16 + hi * 8;
    int b = mrow0 >> 11;            // /T_
    int t0 = mrow0 & (T_ - 1);
    if (z == 2) {                   // V stored transposed [d][t]; rows pack
      Pack8 pk;
      for (int r = 0; r < 8; r++) pk.s[r] = f2bf(acc[i][r]);
      *(uint4*)&VT[((long)(b * H_ + h) * D_ + dim) * T_ + t0] = pk.q;
    } else {
      unsigned short* O = (z == 0) ? Qb : Kb;
      long base = ((long)(b * H_ + h) * T_ + t0) * D_ + dim;
      for (int r = 0; r < 8; r++) O[base + (long)r * D_] = f2bf(acc[i][r]);
    }
  }
}

// --------------------------- pass 1: per-column (query-axis) softmax sums
// grid: (T/64 k-tiles, B*H)  block: 128
__global__ void mha_colsum(const unsigned short* __restrict__ Qb,
                           const unsigned short* __restrict__ Kb,
                           float* __restrict__ COLSUM) {
  __shared__ float lsum[64];
  const int wave = threadIdx.x >> 5;
  const int lane = threadIdx.x & 31;
  const int lo = lane & 15, hi = lane >> 4;
  const int kt = blockIdx.x;
  const int bh = blockIdx.y;
  const int h  = bh & (H_ - 1);
  const float slope = exp2f(-0.5f * (float)(h + 1));  // 2^(-8(h+1)/H)
  const float scale = 0.125f;                          // 1/sqrt(64)

  if (threadIdx.x < 64) lsum[threadIdx.x] = 0.f;
  __syncthreads();

  const unsigned short* Qh = Qb + (long)bh * T_ * D_;
  const unsigned short* Kh = Kb + (long)bh * T_ * D_;
  const int kcol = kt * 64 + wave * 16 + lo;
  float partial = 0.f;
  v8f zero = {};

  for (int qt = kt; qt < T_ / 64; ++qt) {   // causal: only q >= k tiles
    v8f s[4] = {zero, zero, zero, zero};
    for (int kk = 0; kk < D_; kk += 32) {
      Frag16 bf;  // B[d,k] = K[k,d]: column k -> contiguous d in Kb
      const unsigned short* bp = Kh + (long)kcol * D_ + kk + hi * 8;
      bf.q[0] = *(const uint4*)(bp);
      bf.q[1] = *(const uint4*)(bp + 16);
      for (int i = 0; i < 4; i++) {
        Frag16 af;
        const unsigned short* ap = Qh + (long)(qt * 64 + i * 16 + lo) * D_ + kk + hi * 8;
        af.q[0] = *(const uint4*)(ap);
        af.q[1] = *(const uint4*)(ap + 16);
        s[i] = __builtin_amdgcn_wmma_f32_16x16x32_bf16(
            false, af.v, false, bf.v, (short)0, s[i], false, false);
      }
    }
    for (int i = 0; i < 4; i++) {
      int q0 = qt * 64 + i * 16 + hi * 8;
      for (int r = 0; r < 8; r++) {
        int q = q0 + r;
        if (kcol <= q)
          partial += __expf(s[i][r] * scale + (float)(kcol - q) * slope);
      }
    }
  }
  atomicAdd(&lsum[wave * 16 + lo], partial);
  __syncthreads();
  if (threadIdx.x < 64)
    COLSUM[(long)bh * T_ + kt * 64 + threadIdx.x] = lsum[threadIdx.x];
}

// ------------------- pass 2: O[q,d] = sum_k w[q,k] V[k,d], head-summed
// grid: (T/64 q-tiles, B*H)  block: 128
__global__ void mha_av(const unsigned short* __restrict__ Qb,
                       const unsigned short* __restrict__ Kb,
                       const unsigned short* __restrict__ VT,
                       const float* __restrict__ COLSUM,
                       float* __restrict__ OSUM) {       // [B][T][D] fp32
  __shared__ __align__(16) unsigned short Wt[64 * 64];   // [q][k] bf16
  const int wave = threadIdx.x >> 5;
  const int lane = threadIdx.x & 31;
  const int lo = lane & 15, hi = lane >> 4;
  const int qt = blockIdx.x;
  const int bh = blockIdx.y;
  const int b = bh >> 4, h = bh & 15;
  const float slope = exp2f(-0.5f * (float)(h + 1));
  const float scale = 0.125f;

  const unsigned short* Qh = Qb + (long)bh * T_ * D_;
  const unsigned short* Kh = Kb + (long)bh * T_ * D_;
  const unsigned short* Vh = VT + (long)bh * D_ * T_;
  const float* CS = COLSUM + (long)bh * T_;

  v8f zero = {};
  v8f o[4] = {zero, zero, zero, zero};

  for (int kt = 0; kt <= qt; ++kt) {
    // scores for k-strip owned by this wave
    v8f s[4] = {zero, zero, zero, zero};
    const int kcol = kt * 64 + wave * 16 + lo;
    for (int kk = 0; kk < D_; kk += 32) {
      Frag16 bf;
      const unsigned short* bp = Kh + (long)kcol * D_ + kk + hi * 8;
      bf.q[0] = *(const uint4*)(bp);
      bf.q[1] = *(const uint4*)(bp + 16);
      for (int i = 0; i < 4; i++) {
        Frag16 af;
        const unsigned short* ap = Qh + (long)(qt * 64 + i * 16 + lo) * D_ + kk + hi * 8;
        af.q[0] = *(const uint4*)(ap);
        af.q[1] = *(const uint4*)(ap + 16);
        s[i] = __builtin_amdgcn_wmma_f32_16x16x32_bf16(
            false, af.v, false, bf.v, (short)0, s[i], false, false);
      }
    }
    const float rcs = 1.0f / CS[kcol];
    for (int i = 0; i < 4; i++) {
      int q0 = qt * 64 + i * 16 + hi * 8;
      for (int r = 0; r < 8; r++) {
        int q = q0 + r;
        float wv = 0.f;
        if (kcol <= q)
          wv = __expf(s[i][r] * scale + (float)(kcol - q) * slope) * rcs;
        Wt[(i * 16 + hi * 8 + r) * 64 + wave * 16 + lo] = f2bf(wv);
      }
    }
    __syncthreads();
    // O += W * V   (wave owns the 16-wide d-strip wave*16)
    for (int kk = 0; kk < 64; kk += 32) {
      Frag16 bf;  // B[k,d] = V[k,d] contiguous in k via VT[d][t]
      const unsigned short* bp = Vh + (long)(wave * 16 + lo) * T_ + kt * 64 + kk + hi * 8;
      bf.q[0] = *(const uint4*)(bp);
      bf.q[1] = *(const uint4*)(bp + 16);
      for (int i = 0; i < 4; i++) {
        Frag16 af;   // A from LDS: row q, contiguous k chunks
        const unsigned short* ap = &Wt[(i * 16 + lo) * 64 + kk + hi * 8];
        af.q[0] = *(const uint4*)(ap);
        af.q[1] = *(const uint4*)(ap + 16);
        o[i] = __builtin_amdgcn_wmma_f32_16x16x32_bf16(
            false, af.v, false, bf.v, (short)0, o[i], false, false);
      }
    }
    __syncthreads();
  }
  const int dd = wave * 16 + lo;
  for (int i = 0; i < 4; i++) {
    int q0 = qt * 64 + i * 16 + hi * 8;
    for (int r = 0; r < 8; r++)
      atomicAdd(&OSUM[((long)b * T_ + q0 + r) * D_ + dd], o[i][r]);
  }
}

// ------------------------------- final: out = x + OSUM @ wo  (K = 64 only)
__global__ void mha_out(const float* __restrict__ x, const float* __restrict__ wo,
                        const float* __restrict__ OSUM, float* __restrict__ out) {
  long i = (long)blockIdx.x * blockDim.x + threadIdx.x;   // over M_*C_
  int c = (int)(i & (C_ - 1));
  long mt = i >> 10;                                       // b*T + t
  const float* ov = OSUM + mt * D_;
  float acc = 0.f;
  for (int dd = 0; dd < D_; ++dd) acc += ov[dd] * wo[dd * C_ + c];
  out[i] = x[i] + acc;
}

extern "C" void kernel_launch(void* const* d_in, const int* in_sizes, int n_in,
                              void* d_out, int out_size, void* d_ws, size_t ws_size,
                              hipStream_t stream) {
  const float* x  = (const float*)d_in[0];
  const float* wq = (const float*)d_in[1];
  const float* wk = (const float*)d_in[2];
  const float* wv = (const float*)d_in[3];
  const float* wo = (const float*)d_in[4];
  float* out = (float*)d_out;

  char* w = (char*)d_ws;
  size_t off = 0;
  auto alloc = [&](size_t bytes) -> void* {
    void* p = w + off;
    off += (bytes + 255) & ~(size_t)255;
    return p;
  };
  unsigned short* XB  = (unsigned short*)alloc((size_t)M_ * C_ * 2);
  unsigned short* WQT = (unsigned short*)alloc((size_t)C_ * C_ * 2);
  unsigned short* WKT = (unsigned short*)alloc((size_t)C_ * C_ * 2);
  unsigned short* WVT = (unsigned short*)alloc((size_t)C_ * C_ * 2);
  unsigned short* Qb  = (unsigned short*)alloc((size_t)B_ * H_ * T_ * D_ * 2);
  unsigned short* Kb  = (unsigned short*)alloc((size_t)B_ * H_ * T_ * D_ * 2);
  unsigned short* VT  = (unsigned short*)alloc((size_t)B_ * H_ * D_ * T_ * 2);
  float* COLSUM = (float*)alloc((size_t)B_ * H_ * T_ * 4);
  float* OSUM   = (float*)alloc((size_t)B_ * T_ * D_ * 4);

  { int n = B_ * T_ * D_;
    mha_zero<<<(n + 255) / 256, 256, 0, stream>>>(OSUM, n); }
  { long n = (long)M_ * C_ + 3L * C_ * C_;
    mha_convert<<<(int)((n + 255) / 256), 256, 0, stream>>>(x, wq, wk, wv, XB, WQT, WKT, WVT); }
  mha_qkv<<<dim3(M_ / 64, H_, 3), 128, 0, stream>>>(XB, WQT, WKT, WVT, Qb, Kb, VT);
  mha_colsum<<<dim3(T_ / 64, B_ * H_), 128, 0, stream>>>(Qb, Kb, COLSUM);
  mha_av<<<dim3(T_ / 64, B_ * H_), 128, 0, stream>>>(Qb, Kb, VT, COLSUM, OSUM);
  { long n = (long)M_ * C_;
    mha_out<<<(int)(n / 256), 256, 0, stream>>>(x, wo, OSUM, out); }
}
